// PointCloudTeacher_30039001268631
// MI455X (gfx1250) — compile-verified, matrix operand order checked by reading, not run on hardware
//
#include <hip/hip_runtime.h>
#include <hip/hip_bf16.h>
#include <cstdint>
#include <cstddef>

// ---------------------------------------------------------------------------
// Types for CDNA5 WMMA (wave32): v_wmma_f32_16x16x32_bf16
// ---------------------------------------------------------------------------
typedef __bf16 bf16_t;
typedef __attribute__((ext_vector_type(8)))  __bf16 v8bf;
typedef __attribute__((ext_vector_type(16))) __bf16 v16bf;
typedef __attribute__((ext_vector_type(8)))  float  v8f;

__device__ __forceinline__ bf16_t f2bf(float f) {
  uint32_t u = __float_as_uint(f);
  uint32_t r = (u + 0x7FFFu + ((u >> 16) & 1u)) >> 16;   // round-to-nearest-even
  return __builtin_bit_cast(bf16_t, (uint16_t)r);
}

// One lane's 16 bf16 fragment elements for a 32-wide K step.
// 16-bit 16x32 A layout (mirrored for 32x16 B): lane%16 selects row (A) /
// column (B); lane/16 selects K-halves {0..7,16..23} vs {8..15,24..31}.
// Caller pre-offsets by (lane>>4)*8, so each lane does two 16B loads (+0,+16).
__device__ __forceinline__ v16bf ld16(const bf16_t* p) {
  v8bf lo = *reinterpret_cast<const v8bf*>(p);
  v8bf hi = *reinterpret_cast<const v8bf*>(p + 16);
  v16bf r;
#pragma unroll
  for (int i = 0; i < 8; ++i) { r[i] = lo[i]; r[i + 8] = hi[i]; }
  return r;
}

__device__ __forceinline__ v8f wmma_bf16(v16bf a, v16bf b, v8f c) {
  return __builtin_amdgcn_wmma_f32_16x16x32_bf16(false, a, false, b, (short)0, c,
                                                 false, false);
}

// ---------------------------------------------------------------------------
// Kernel 1: per-row fp32 -> bf16 convert + squared-norm (one wave per row)
// ---------------------------------------------------------------------------
__launch_bounds__(256)
__global__ void k_rows_bf16_sqn(const float* __restrict__ X, bf16_t* __restrict__ Xb,
                                float* __restrict__ sqn, int C) {
  int row  = blockIdx.x * 8 + (threadIdx.x >> 5);
  int lane = threadIdx.x & 31;
  const float* src = X + (size_t)row * C;
  bf16_t* dst = Xb + (size_t)row * C;
  float ss = 0.f;
  for (int i = lane; i < C; i += 32) {
    float v = src[i];
    ss += v * v;
    dst[i] = f2bf(v);
  }
#pragma unroll
  for (int off = 16; off > 0; off >>= 1) ss += __shfl_down(ss, off, 32);
  if (lane == 0) sqn[row] = ss;
}

// ---------------------------------------------------------------------------
// Kernel 2: weight prep.  W:(H,2C) -> Wl[o,i]=W[o,i], Wd[o,i]=W[o,C+i]-W[o,i]
// ---------------------------------------------------------------------------
__launch_bounds__(256)
__global__ void k_prep_w(const float* __restrict__ W, bf16_t* __restrict__ Wl,
                         bf16_t* __restrict__ Wd, int H, int C) {
  size_t t = (size_t)blockIdx.x * blockDim.x + threadIdx.x;
  if (t >= (size_t)H * C) return;
  int o = (int)(t / C), i = (int)(t % C);
  float wl = W[(size_t)o * (2 * C) + i];
  float wr = W[(size_t)o * (2 * C) + C + i];
  Wl[t] = f2bf(wl);
  Wd[t] = f2bf(wr - wl);
}

// ---------------------------------------------------------------------------
// Kernel 3: KNN top-5 of (2*dot - |x_m|^2) per row via WMMA gram tiles.
// Grid: (N/16, B); block 128 (4 waves).  A strip (16 x C) staged in LDS.
// Each wave processes column-tile PAIRS (16x32 of scores per iteration) with
// a software-pipelined (double-buffered) K loop: the kt+1 fragments are in
// flight while the kt WMMAs execute, so the pre-WMMA waits no longer drain
// the counters to zero.
// ---------------------------------------------------------------------------
__launch_bounds__(128)
__global__ void k_knn(const bf16_t* __restrict__ Xb, const float* __restrict__ sqn,
                      int* __restrict__ idx_out, int N, int C) {
  extern __shared__ char smem[];
  bf16_t* As = (bf16_t*)smem;                         // 16*C bf16
  float*  Sc = (float*)(smem + (size_t)16 * C * 2);   // 4 waves x 16x32 scores
  float*  Mv = Sc + 4 * 16 * 32;                      // 4 x 16 x 5 merge vals
  int*    Mi = (int*)(Mv + 4 * 16 * 5);               // 4 x 16 x 5 merge idx

  int b       = blockIdx.y;
  int rowbase = blockIdx.x * 16;
  int tid  = threadIdx.x;
  int wave = tid >> 5, lane = tid & 31;
  int ln16 = lane & 15, kh = lane >> 4;

  const bf16_t* X  = Xb  + (size_t)b * N * C;
  const float*  sq = sqn + (size_t)b * N;

  // cooperative copy of the contiguous 16-row A strip into LDS
  {
    const uint32_t* src = (const uint32_t*)(X + (size_t)rowbase * C);
    uint32_t* dst = (uint32_t*)As;
    int nw = 8 * C;                                   // 16*C*2 bytes / 4
    for (int i = tid; i < nw; i += 128) dst[i] = src[i];
  }
  __syncthreads();

  float tv[5]; int ti[5];
#pragma unroll
  for (int k = 0; k < 5; ++k) { tv[k] = -3.0e38f; ti[k] = 0; }

  const bf16_t* ap = As + (size_t)ln16 * C + kh * 8;
  int nkt    = C >> 5;
  int npairs = N >> 5;                                // 128; 32 pairs per wave

  for (int p = wave; p < npairs; p += 4) {
    int colbase0 = p << 5;
    int colbase1 = colbase0 + 16;
    const bf16_t* bp0 = X + (size_t)(colbase0 + ln16) * C + kh * 8;
    const bf16_t* bp1 = bp0 + (size_t)16 * C;

    v8f acc0 = {}, acc1 = {};
    v16bf a  = ld16(ap);
    v16bf b0 = ld16(bp0);
    v16bf b1 = ld16(bp1);
    for (int kt = 0; kt < nkt - 1; ++kt) {
      __builtin_prefetch(bp0 + (kt + 5) * 32, 0, 1);
      __builtin_prefetch(bp1 + (kt + 5) * 32, 0, 1);
      v16bf an  = ld16(ap  + (kt + 1) * 32);
      v16bf b0n = ld16(bp0 + (kt + 1) * 32);
      v16bf b1n = ld16(bp1 + (kt + 1) * 32);
      acc0 = wmma_bf16(a, b0, acc0);
      acc1 = wmma_bf16(a, b1, acc1);
      a = an; b0 = b0n; b1 = b1n;
    }
    acc0 = wmma_bf16(a, b0, acc0);
    acc1 = wmma_bf16(a, b1, acc1);

    // score = 2*dot - |x_col|^2 (row-constant term dropped; order-preserving)
    float  sqc0 = sq[colbase0 + ln16];
    float  sqc1 = sq[colbase1 + ln16];
    float* sw   = Sc + wave * 512;
#pragma unroll
    for (int r = 0; r < 8; ++r) {
      int m = (r + 8 * kh) * 32;                      // C/D: M=r+8*(lane/16), N=lane%16
      sw[m + ln16]      = 2.f * acc0[r] - sqc0;
      sw[m + 16 + ln16] = 2.f * acc1[r] - sqc1;
    }
    __syncthreads();
    if (lane < 16) {
      const float* srow = Sc + wave * 512 + lane * 32;
#pragma unroll
      for (int c = 0; c < 32; ++c) {
        float v = srow[c];
        if (v > tv[4]) {
          int j = 4;
          while (j > 0 && tv[j - 1] < v) { tv[j] = tv[j - 1]; ti[j] = ti[j - 1]; --j; }
          tv[j] = v; ti[j] = colbase0 + c;
        }
      }
    }
    __syncthreads();
  }

  // cross-wave merge of per-wave top-5 lists
  if (lane < 16) {
#pragma unroll
    for (int k = 0; k < 5; ++k) {
      Mv[(wave * 16 + lane) * 5 + k] = tv[k];
      Mi[(wave * 16 + lane) * 5 + k] = ti[k];
    }
  }
  __syncthreads();
  if (wave == 0 && lane < 16) {
    float fv[5]; int fi[5];
#pragma unroll
    for (int k = 0; k < 5; ++k) { fv[k] = -3.0e38f; fi[k] = 0; }
    for (int w = 0; w < 4; ++w)
      for (int k = 0; k < 5; ++k) {
        float v  = Mv[(w * 16 + lane) * 5 + k];
        int   id = Mi[(w * 16 + lane) * 5 + k];
        if (v > fv[4]) {
          int j = 4;
          while (j > 0 && fv[j - 1] < v) { fv[j] = fv[j - 1]; fi[j] = fi[j - 1]; --j; }
          fv[j] = v; fi[j] = id;
        }
      }
    int* op = idx_out + ((size_t)b * N + rowbase + lane) * 5;
#pragma unroll
    for (int k = 0; k < 5; ++k) op[k] = fi[k];
  }
}

// ---------------------------------------------------------------------------
// Kernel 4: Y = X(BNxC,bf16) * W(HxC,bf16)^T -> f32 (BNxH).
// One wave per 16x32 output tile-pair (A fragment reused by two WMMAs),
// double-buffered K loop.  8 waves/block share the same 16 A-rows (L0 reuse).
// ---------------------------------------------------------------------------
__launch_bounds__(256)
__global__ void k_gemm(const bf16_t* __restrict__ X, const bf16_t* __restrict__ W,
                       float* __restrict__ Y, int C, int H) {
  int wave = threadIdx.x >> 5, lane = threadIdx.x & 31;
  int ln16 = lane & 15, kh = lane >> 4;
  int pair   = blockIdx.x * 8 + wave;
  int npairs = H >> 5;                          // H/16 tiles, processed in pairs
  if (pair >= npairs) return;                   // wave-uniform: EXEC stays all-1s
  int colt0 = pair << 1;
  size_t rowbase = (size_t)blockIdx.y * 16;

  const bf16_t* xp  = X + (rowbase + ln16) * C + kh * 8;
  const bf16_t* wp0 = W + ((size_t)colt0 * 16 + ln16) * C + kh * 8;
  const bf16_t* wp1 = wp0 + (size_t)16 * C;

  v8f acc0 = {}, acc1 = {};
  int nkt = C >> 5;
  v16bf a  = ld16(xp);
  v16bf b0 = ld16(wp0);
  v16bf b1 = ld16(wp1);
  for (int kt = 0; kt < nkt - 1; ++kt) {
    __builtin_prefetch(xp  + (kt + 5) * 32, 0, 1);
    __builtin_prefetch(wp0 + (kt + 5) * 32, 0, 1);
    __builtin_prefetch(wp1 + (kt + 5) * 32, 0, 1);
    v16bf an  = ld16(xp  + (kt + 1) * 32);
    v16bf b0n = ld16(wp0 + (kt + 1) * 32);
    v16bf b1n = ld16(wp1 + (kt + 1) * 32);
    acc0 = wmma_bf16(a, b0, acc0);
    acc1 = wmma_bf16(a, b1, acc1);
    a = an; b0 = b0n; b1 = b1n;
  }
  acc0 = wmma_bf16(a, b0, acc0);
  acc1 = wmma_bf16(a, b1, acc1);

  float* yp = Y + rowbase * H + colt0 * 16 + ln16;
  int mb = 8 * kh;
#pragma unroll
  for (int r = 0; r < 8; ++r) {
    yp[(size_t)(mb + r) * H]      = acc0[r];
    yp[(size_t)(mb + r) * H + 16] = acc1[r];
  }
}

// ---------------------------------------------------------------------------
// Kernel 5: out[b,n,o] = lrelu( max_k ((Y[b,idx_k,o]+Z[b,n,o]-mean)*scale+beta) )
// float4-vectorized over channels (H % 4 == 0, rows 16B-aligned); the 5 index
// loads amortize over 4 channels.  LeakyReLU commutes with the outer max
// (monotone); BN affine applied per-k since scale may be negative.
// ---------------------------------------------------------------------------
__launch_bounds__(256)
__global__ void k_edge_reduce(const float* __restrict__ Y, const float* __restrict__ Z,
                              const int* __restrict__ idx,
                              const float* __restrict__ gamma, const float* __restrict__ beta,
                              const float* __restrict__ mean,  const float* __restrict__ var,
                              float* __restrict__ out, int N, int H, long long total4) {
  size_t t = (size_t)blockIdx.x * blockDim.x + threadIdx.x;
  if (t >= (size_t)total4) return;
  int    H4 = H >> 2;
  int    o  = (int)(t % H4) << 2;
  size_t bn = t / H4;
  int    b  = (int)(bn / N);
  const int* id = idx + bn * 5;

  float4 ga = *(const float4*)(gamma + o);
  float4 be = *(const float4*)(beta + o);
  float4 mu = *(const float4*)(mean + o);
  float4 va = *(const float4*)(var + o);
  float4 sc, sh;
  sc.x = ga.x * rsqrtf(va.x + 1e-5f); sh.x = be.x - mu.x * sc.x;
  sc.y = ga.y * rsqrtf(va.y + 1e-5f); sh.y = be.y - mu.y * sc.y;
  sc.z = ga.z * rsqrtf(va.z + 1e-5f); sh.z = be.z - mu.z * sc.z;
  sc.w = ga.w * rsqrtf(va.w + 1e-5f); sh.w = be.w - mu.w * sc.w;

  float4 zc = *(const float4*)(Z + bn * H + o);
  float4 m  = make_float4(-3.0e38f, -3.0e38f, -3.0e38f, -3.0e38f);
#pragma unroll
  for (int k = 0; k < 5; ++k) {
    float4 y = *(const float4*)(Y + ((size_t)b * N + id[k]) * H + o);
    m.x = fmaxf(m.x, (y.x + zc.x) * sc.x + sh.x);
    m.y = fmaxf(m.y, (y.y + zc.y) * sc.y + sh.y);
    m.z = fmaxf(m.z, (y.z + zc.z) * sc.z + sh.z);
    m.w = fmaxf(m.w, (y.w + zc.w) * sc.w + sh.w);
  }
  float4 r;
  r.x = (m.x >= 0.f) ? m.x : 0.2f * m.x;
  r.y = (m.y >= 0.f) ? m.y : 0.2f * m.y;
  r.z = (m.z >= 0.f) ? m.z : 0.2f * m.z;
  r.w = (m.w >= 0.f) ? m.w : 0.2f * m.w;
  *(float4*)(out + bn * H + o) = r;
}

// ---------------------------------------------------------------------------
// Host orchestration
// ---------------------------------------------------------------------------
extern "C" void kernel_launch(void* const* d_in, const int* in_sizes, int n_in,
                              void* d_out, int out_size, void* d_ws, size_t ws_size,
                              hipStream_t stream) {
  (void)in_sizes; (void)n_in; (void)out_size; (void)ws_size;
  const int B = 4, N = 4096, C1 = 512, H1 = 864, C2 = 864, H2 = 1728;
  const size_t BN = (size_t)B * N;

  const float* X0 = (const float*)d_in[0];
  const float* W1 = (const float*)d_in[1];
  const float* g1 = (const float*)d_in[2];
  const float* b1 = (const float*)d_in[3];
  const float* m1 = (const float*)d_in[4];
  const float* v1 = (const float*)d_in[5];
  const float* W2 = (const float*)d_in[6];
  const float* g2 = (const float*)d_in[7];
  const float* b2 = (const float*)d_in[8];
  const float* m2 = (const float*)d_in[9];
  const float* v2 = (const float*)d_in[10];

  // bump allocator over d_ws (deterministic, graph-capture safe)
  char* wsb = (char*)d_ws;
  size_t off = 0;
  auto alloc = [&](size_t bytes) -> char* {
    off = (off + 255) & ~(size_t)255;
    char* p = wsb + off;
    off += bytes;
    return p;
  };

  float* X1 = (float*)alloc(BN * H1 * 4);      // persistent block-1 output
  size_t mark = off;                           // everything below is reusable

  // ---- block 1 scratch
  bf16_t* xb1 = (bf16_t*)alloc(BN * C1 * 2);
  float*  sq1 = (float*) alloc(BN * 4);
  int*    id1 = (int*)   alloc(BN * 5 * 4);
  bf16_t* wl1 = (bf16_t*)alloc((size_t)H1 * C1 * 2);
  bf16_t* wd1 = (bf16_t*)alloc((size_t)H1 * C1 * 2);
  float*  Y1  = (float*) alloc(BN * H1 * 4);
  float*  Z1  = (float*) alloc(BN * H1 * 4);

  // ---- block 1
  k_rows_bf16_sqn<<<(int)(BN / 8), 256, 0, stream>>>(X0, xb1, sq1, C1);
  k_prep_w<<<(int)(((size_t)H1 * C1 + 255) / 256), 256, 0, stream>>>(W1, wl1, wd1, H1, C1);
  {
    dim3 gk(N / 16, B);
    size_t lds = (size_t)16 * C1 * 2 + 4 * 16 * 32 * 4 + 4 * 16 * 5 * 8;
    k_knn<<<gk, 128, lds, stream>>>(xb1, sq1, id1, N, C1);
  }
  {
    dim3 gg(((H1 >> 5) + 7) / 8, (int)(BN / 16));
    k_gemm<<<gg, 256, 0, stream>>>(xb1, wl1, Y1, C1, H1);
    k_gemm<<<gg, 256, 0, stream>>>(xb1, wd1, Z1, C1, H1);
  }
  {
    long long tot4 = (long long)BN * (H1 >> 2);
    k_edge_reduce<<<(int)((tot4 + 255) / 256), 256, 0, stream>>>(
        Y1, Z1, id1, g1, b1, m1, v1, X1, N, H1, tot4);
  }

  // ---- block 2 scratch (overlays block-1 scratch; stream-ordered so safe)
  off = mark;
  bf16_t* xb2 = (bf16_t*)alloc(BN * C2 * 2);
  float*  sq2 = (float*) alloc(BN * 4);
  int*    id2 = (int*)   alloc(BN * 5 * 4);
  bf16_t* wl2 = (bf16_t*)alloc((size_t)H2 * C2 * 2);
  bf16_t* wd2 = (bf16_t*)alloc((size_t)H2 * C2 * 2);
  float*  Y2  = (float*) alloc(BN * H2 * 4);
  float*  Z2  = (float*) alloc(BN * H2 * 4);

  // ---- block 2
  k_rows_bf16_sqn<<<(int)(BN / 8), 256, 0, stream>>>(X1, xb2, sq2, C2);
  k_prep_w<<<(int)(((size_t)H2 * C2 + 255) / 256), 256, 0, stream>>>(W2, wl2, wd2, H2, C2);
  {
    dim3 gk(N / 16, B);
    size_t lds = (size_t)16 * C2 * 2 + 4 * 16 * 32 * 4 + 4 * 16 * 5 * 8;
    k_knn<<<gk, 128, lds, stream>>>(xb2, sq2, id2, N, C2);
  }
  {
    dim3 gg(((H2 >> 5) + 7) / 8, (int)(BN / 16));
    k_gemm<<<gg, 256, 0, stream>>>(xb2, wl2, Y2, C2, H2);
    k_gemm<<<gg, 256, 0, stream>>>(xb2, wd2, Z2, C2, H2);
  }
  {
    long long tot4 = (long long)BN * (H2 >> 2);
    k_edge_reduce<<<(int)((tot4 + 255) / 256), 256, 0, stream>>>(
        Y2, Z2, id2, g2, b2, m2, v2, (float*)d_out, N, H2, tot4);
  }
}
